// NeuralODEDecoder_30983894073712
// MI455X (gfx1250) — matrix operand order repaired and checked by reading, not compiled, not run on hardware
//
#include <hip/hip_runtime.h>
#include <hip/hip_bf16.h>
#include <math.h>

// ---- problem constants (match reference) ----
#define BB   4096
#define LL   256
#define HH   512
#define OO   128
#define TT   8
#define NL   5
#define ROWS 32          // batch rows per workgroup (2 M-tiles, in-wave B reuse)
#define NTHREADS 256     // 8 waves (wave32)

typedef __attribute__((ext_vector_type(16))) __bf16 v16bf;
typedef __attribute__((ext_vector_type(8)))  __bf16 v8bf;
typedef __attribute__((ext_vector_type(8)))  float  v8f;

__device__ __forceinline__ float eluf(float x) { return x > 0.f ? x : expm1f(x); }

// ---------------------------------------------------------------------------
// GEMM core: D[2x16, NT*16] += A[32,K](bf16, LDS, stride 512) * W^T
// W is global bf16, row-major [Nout][K] (output-row-major, as in x @ W.T).
// Each B fragment is loaded once and reused for both M-tiles (2x intensity).
// ---------------------------------------------------------------------------
template<int K, int NT>
__device__ __forceinline__ void wmma_gemm(const __bf16* __restrict__ sAbf,
                                          const __bf16* __restrict__ W,
                                          int ntile0, int lane, v8f (&c)[2][NT]) {
  const int r   = lane & 15;
  const int hi  = lane >> 4;
  const int kaA = hi << 4;   // +0 / +16 for A fragment K base
  const int kaB = hi << 4;   // +0 / +16 for B fragment K base
#pragma unroll
  for (int kk = 0; kk < K; kk += 32) {
    union { v16bf v; v8bf h[2]; } a0, a1;
    const __bf16* ap0 = sAbf + r * 512 + kk + kaA;           // M-tile 0
    a0.h[0] = *(const v8bf*)(ap0);
    a0.h[1] = *(const v8bf*)(ap0 + 8);
    const __bf16* ap1 = ap0 + 16 * 512;                      // M-tile 1
    a1.h[0] = *(const v8bf*)(ap1);
    a1.h[1] = *(const v8bf*)(ap1 + 8);
#pragma unroll
    for (int nt = 0; nt < NT; ++nt) {
      const int col = (ntile0 + nt) * 16 + r;
      const v16bf b = *(const v16bf*)(W + col * K + kk + kaB);
      c[0][nt] = __builtin_amdgcn_wmma_f32_16x16x32_bf16(
                     false, a0.v, false, b, (short)0, c[0][nt], false, false);
      c[1][nt] = __builtin_amdgcn_wmma_f32_16x16x32_bf16(
                     false, a1.v, false, b, (short)0, c[1][nt], false, false);
    }
  }
}

// ---------------------------------------------------------------------------
// LayerNorm over n (256/512) cols of a [32, srcStride] f32 LDS tile,
// writing bf16 into dst (stride 512). 8 threads per row, shfl reductions
// (xor masks 1,2,4 stay inside each 8-lane group on wave32).
// ---------------------------------------------------------------------------
__device__ __forceinline__ void ln_to_bf(const float* __restrict__ src, int srcStride,
                                         int n, const float* __restrict__ g,
                                         const float* __restrict__ bta,
                                         __bf16* __restrict__ dst, int tid) {
  const int row = tid >> 3;        // 0..31
  const int sub = tid & 7;         // 0..7
  const int per = n >> 3;          // 64 (n=512) or 32 (n=256)
  const float* p = src + row * srcStride + sub * per;
  float s = 0.f, s2 = 0.f;
  for (int j = 0; j < per; ++j) { float x = p[j]; s += x; s2 += x * x; }
#pragma unroll
  for (int m = 1; m <= 4; m <<= 1) {
    s  += __shfl_xor(s,  m, 32);
    s2 += __shfl_xor(s2, m, 32);
  }
  const float mean = s / (float)n;
  const float var  = fmaxf(s2 / (float)n - mean * mean, 0.f);
  const float inv  = rsqrtf(var + 1e-5f);
  __bf16* q = dst + row * 512 + sub * per;
  for (int j = 0; j < per; ++j) {
    const int col = sub * per + j;
    q[j] = (__bf16)(((p[j] - mean) * inv) * g[col] + bta[col]);
  }
}

// ---------------------------------------------------------------------------
// f32 -> bf16 weight conversion
// ---------------------------------------------------------------------------
__global__ void cvt_f32_bf16(const float* __restrict__ src, __bf16* __restrict__ dst, int n) {
  int i = blockIdx.x * 256 + threadIdx.x;
  if (i < n) dst[i] = (__bf16)src[i];
}

// ---------------------------------------------------------------------------
// Fully fused Neural-ODE + decoder. One workgroup owns 32 batch rows for the
// entire time loop; all activations/RK4 state live in LDS (288 KB / WGP).
// ---------------------------------------------------------------------------
__global__ __launch_bounds__(NTHREADS)
void node_fused(const float* __restrict__ z0, const float* __restrict__ tg,
                const __bf16* __restrict__ Winb, const float* __restrict__ b_in,
                const float* __restrict__ mlp_g, const float* __restrict__ mlp_b,
                const __bf16* __restrict__ mlpWb, const float* __restrict__ mlp_bb,
                const float* __restrict__ ode_w,
                const float* __restrict__ fn_g, const float* __restrict__ fn_b,
                const __bf16* __restrict__ Woutb, const float* __restrict__ b_out,
                const float* __restrict__ dn_g, const float* __restrict__ dn_b,
                const __bf16* __restrict__ W1b, const float* __restrict__ b1,
                const __bf16* __restrict__ W2b, const float* __restrict__ b2,
                float* __restrict__ out) {
  __shared__ __bf16 sA [ROWS * 512];   // 32 KB  GEMM A staging (bf16)
  __shared__ __bf16 sA2[ROWS * 512];   // 32 KB  decoder hidden (bf16)
  __shared__ float  sH  [ROWS * 512];  // 64 KB  layer activations (f32)
  __shared__ float  sAcc[ROWS * 512];  // 64 KB  ode weighted sum (f32)
  __shared__ float  sZ   [ROWS * 256]; // 32 KB  RK4 z
  __shared__ float  sZarg[ROWS * 256]; // 32 KB  RK4 eval point
  __shared__ float  sKsum[ROWS * 256]; // 32 KB  RK4 k accumulator

  const int tid  = threadIdx.x;
  const int wave = tid >> 5;
  const int lane = tid & 31;
  const int r    = lane & 15;
  const int hi8  = (lane >> 4) * 8;
  const int r0   = blockIdx.x * ROWS;

  float ow[NL + 1];
#pragma unroll
  for (int j = 0; j < NL + 1; ++j) ow[j] = ode_w[j];

  // ---- load z0 tile ----
  for (int i = tid; i < ROWS * 256; i += NTHREADS) {
    const int row = i >> 8, col = i & 255;
    const float z = z0[(size_t)(r0 + row) * 256 + col];
    sZ[i] = z; sZarg[i] = z;
  }
  __syncthreads();

  // ---- decoder: xs[t] = elu(LN(z) @ W1^T + b1) @ W2^T + b2 ----
  auto decoder = [&](int tstep) {
    ln_to_bf(sZ, 256, 256, dn_g, dn_b, sA, tid);
    __syncthreads();
    v8f c1[2][4] = {};
    wmma_gemm<256, 4>(sA, W1b, wave * 4, lane, c1);
#pragma unroll
    for (int m = 0; m < 2; ++m)
#pragma unroll
      for (int nt = 0; nt < 4; ++nt)
#pragma unroll
        for (int v = 0; v < 8; ++v) {
          const int col = (wave * 4 + nt) * 16 + r;
          const int row = m * 16 + v + hi8;
          sA2[row * 512 + col] = (__bf16)eluf(c1[m][nt][v] + b1[col]);
        }
    __syncthreads();
    v8f c2[2][1] = {};
    wmma_gemm<512, 1>(sA2, W2b, wave, lane, c2);
#pragma unroll
    for (int m = 0; m < 2; ++m)
#pragma unroll
      for (int v = 0; v < 8; ++v) {
        const int col = wave * 16 + r;
        const int row = m * 16 + v + hi8;
        out[(size_t)tstep * BB * OO + (size_t)(r0 + row) * OO + col] = c2[m][0][v] + b2[col];
      }
    __syncthreads();
  };

  // ---- one evaluation of f(zarg) with RK4 stage update ----
  auto feval = [&](int stage, float dt) {
    // stage zarg -> bf16
    for (int i = tid; i < ROWS * 256; i += NTHREADS)
      sA[(i >> 8) * 512 + (i & 255)] = (__bf16)sZarg[i];
    __syncthreads();

    // h = elu(z @ Win^T + b_in); acc = w0*h
    {
      v8f c[2][4] = {};
      wmma_gemm<256, 4>(sA, Winb, wave * 4, lane, c);
#pragma unroll
      for (int m = 0; m < 2; ++m)
#pragma unroll
        for (int nt = 0; nt < 4; ++nt)
#pragma unroll
          for (int v = 0; v < 8; ++v) {
            const int col = (wave * 4 + nt) * 16 + r;
            const int row = m * 16 + v + hi8;
            const float e = eluf(c[m][nt][v] + b_in[col]);
            sH[row * 512 + col]   = e;
            sAcc[row * 512 + col] = ow[0] * e;
          }
    }
    __syncthreads();

    // 5 x (LN -> Linear -> ELU), acc += w[i+1]*h
    for (int il = 0; il < NL; ++il) {
      ln_to_bf(sH, 512, 512, mlp_g + il * 512, mlp_b + il * 512, sA, tid);
      __syncthreads();
      v8f c[2][4] = {};
      wmma_gemm<512, 4>(sA, mlpWb + (size_t)il * 512 * 512, wave * 4, lane, c);
      const float* bb = mlp_bb + il * 512;
#pragma unroll
      for (int m = 0; m < 2; ++m)
#pragma unroll
        for (int nt = 0; nt < 4; ++nt)
#pragma unroll
          for (int v = 0; v < 8; ++v) {
            const int col = (wave * 4 + nt) * 16 + r;
            const int row = m * 16 + v + hi8;
            const float e = eluf(c[m][nt][v] + bb[col]);
            sH[row * 512 + col]    = e;
            sAcc[row * 512 + col] += ow[il + 1] * e;
          }
      __syncthreads();
    }

    // k = LN(acc) @ Wout^T + b_out, fused RK4 stage update
    ln_to_bf(sAcc, 512, 512, fn_g, fn_b, sA, tid);
    __syncthreads();
    v8f c[2][2] = {};
    wmma_gemm<512, 2>(sA, Woutb, wave * 2, lane, c);
    const float halfdt = 0.5f * dt;
#pragma unroll
    for (int m = 0; m < 2; ++m)
#pragma unroll
      for (int nt = 0; nt < 2; ++nt)
#pragma unroll
        for (int v = 0; v < 8; ++v) {
          const int col = (wave * 2 + nt) * 16 + r;
          const int row = m * 16 + v + hi8;
          const int idx = row * 256 + col;
          const float k = c[m][nt][v] + b_out[col];
          if (stage == 0) { sKsum[idx] = k;             sZarg[idx] = sZ[idx] + halfdt * k; }
          else if (stage == 1) { sKsum[idx] += 2.f * k; sZarg[idx] = sZ[idx] + halfdt * k; }
          else if (stage == 2) { sKsum[idx] += 2.f * k; sZarg[idx] = sZ[idx] + dt * k; }
          else { const float zn = sZ[idx] + (dt / 6.f) * (sKsum[idx] + k);
                 sZ[idx] = zn; sZarg[idx] = zn; }
        }
    __syncthreads();
  };

  decoder(0);
  for (int iv = 0; iv < TT - 1; ++iv) {
    const float dt = (tg[iv + 1] - tg[iv]) * 0.5f;   // / N_SUB
    for (int sub = 0; sub < 2; ++sub)
      for (int st = 0; st < 4; ++st)
        feval(st, dt);
    decoder(iv + 1);
  }
}

// ---------------------------------------------------------------------------
extern "C" void kernel_launch(void* const* d_in, const int* in_sizes, int n_in,
                              void* d_out, int out_size, void* d_ws, size_t ws_size,
                              hipStream_t stream) {
  const float* z0     = (const float*)d_in[0];
  const float* tg     = (const float*)d_in[1];
  const float* W_in   = (const float*)d_in[2];
  const float* b_in   = (const float*)d_in[3];
  const float* mlp_g  = (const float*)d_in[4];
  const float* mlp_b  = (const float*)d_in[5];
  const float* mlp_W  = (const float*)d_in[6];
  const float* mlp_bb = (const float*)d_in[7];
  const float* ode_w  = (const float*)d_in[8];
  const float* fn_g   = (const float*)d_in[9];
  const float* fn_b   = (const float*)d_in[10];
  const float* W_out  = (const float*)d_in[11];
  const float* b_out  = (const float*)d_in[12];
  const float* dn_g   = (const float*)d_in[13];
  const float* dn_b   = (const float*)d_in[14];
  const float* W1     = (const float*)d_in[15];
  const float* b1     = (const float*)d_in[16];
  const float* W2     = (const float*)d_in[17];
  const float* b2     = (const float*)d_in[18];
  float* out = (float*)d_out;

  // bf16 weight staging in workspace
  __bf16* p = (__bf16*)d_ws;
  __bf16* Winb  = p; p += (size_t)HH * LL;        // 512*256
  __bf16* mlpWb = p; p += (size_t)NL * HH * HH;   // 5*512*512
  __bf16* Woutb = p; p += (size_t)LL * HH;        // 256*512
  __bf16* W1b   = p; p += (size_t)HH * LL;        // 512*256
  __bf16* W2b   = p; p += (size_t)OO * HH;        // 128*512

  auto cvt = [&](const float* s, __bf16* d, int n) {
    cvt_f32_bf16<<<(n + 255) / 256, 256, 0, stream>>>(s, d, n);
  };
  cvt(W_in,  Winb,  HH * LL);
  cvt(mlp_W, mlpWb, NL * HH * HH);
  cvt(W_out, Woutb, LL * HH);
  cvt(W1,    W1b,   HH * LL);
  cvt(W2,    W2b,   OO * HH);

  node_fused<<<BB / ROWS, NTHREADS, 0, stream>>>(
      z0, tg, Winb, b_in, mlp_g, mlp_b, mlpWb, mlp_bb, ode_w,
      fn_g, fn_b, Woutb, b_out, dn_g, dn_b, W1b, b1, W2b, b2, out);
}
